// MHA_13537736917614
// MI455X (gfx1250) — matrix operand order repaired and checked by reading, not compiled, not run on hardware
//
#include <hip/hip_runtime.h>

typedef __bf16 v16bf __attribute__((ext_vector_type(16)));
typedef float v8f __attribute__((ext_vector_type(8)));
typedef unsigned int u32x4 __attribute__((ext_vector_type(4)));
typedef unsigned short u16;

union FragU {
  v16bf v;
  u32x4 q[2];
};

__device__ __forceinline__ u16 f32_to_bf16(float f) {
  unsigned int u = __float_as_uint(f);
  u += 0x7FFFu + ((u >> 16) & 1u);  // round-to-nearest-even
  return (u16)(u >> 16);
}

__device__ __forceinline__ v8f vzero8() {
  v8f z = {0.f, 0.f, 0.f, 0.f, 0.f, 0.f, 0.f, 0.f};
  return z;
}

// A fragment (16x32 bf16, MxK): lane holds row M = lane%16,
// K = (lane/16)*8 + {0..7} and +16..+23  -> two contiguous 16B chunks.
__device__ __forceinline__ v16bf load_fragA(const u16* base, int ld, int m0,
                                            int k0, int lane) {
  const u16* p = base + (size_t)(m0 + (lane & 15)) * ld + k0 + ((lane >> 4) << 3);
  FragU f;
  f.q[0] = *reinterpret_cast<const u32x4*>(p);
  f.q[1] = *reinterpret_cast<const u32x4*>(p + 16);
  return f.v;
}

// B fragment (32x16 bf16, KxN): lane holds column N = n0 + lane%16,
// K = (lane/16)*8 + {0..7} and +16..+23 -> two contiguous 16B chunks from "row n".
__device__ __forceinline__ v16bf load_fragB(const u16* base, int ld, int n0,
                                            int k0, int lane) {
  const u16* p = base + (size_t)(n0 + (lane & 15)) * ld + k0 + ((lane >> 4) << 3);
  FragU f;
  f.q[0] = *reinterpret_cast<const u32x4*>(p);
  f.q[1] = *reinterpret_cast<const u32x4*>(p + 16);
  return f.v;
}

__device__ __forceinline__ v8f wmma_bf16(v16bf a, v16bf b, v8f c) {
  return __builtin_amdgcn_wmma_f32_16x16x32_bf16(false, a, false, b, (short)0,
                                                 c, false, false);
}

// ---- CDNA5 async global->LDS copy (ASYNCcnt-tracked DMA-style path) ----
__device__ __forceinline__ void async_ld_b128(unsigned lds_off,
                                              unsigned long long gaddr) {
  asm volatile("global_load_async_to_lds_b128 %0, %1, off" ::"v"(lds_off),
               "v"(gaddr)
               : "memory");
}
__device__ __forceinline__ void wait_async_le4() {
  asm volatile("s_wait_asynccnt 0x4" ::: "memory");
}
__device__ __forceinline__ void wait_async_0() {
  asm volatile("s_wait_asynccnt 0x0" ::: "memory");
}

// ---------------------------------------------------------------- convert
__global__ __launch_bounds__(256) void cvt_bf16_k(const float* __restrict__ in,
                                                  u16* __restrict__ out, int n) {
  int i = blockIdx.x * 256 + threadIdx.x;
  if (i < n) out[i] = f32_to_bf16(in[i]);
}

// ---------------------------------------------------------------- GEMM
// C[m,n] = sum_k A[m,k] * W[n,k]   (A: 4096x2048 bf16, W: 2048x2048 bf16)
// Each wave: 32x64 C tile, K step 32 -> 8 v_wmma per step, 512 per wave.
// All 6 fragments for a K-step are hoisted into registers so the
// global_load_b128 clause runs ahead of the WMMA chain.
// MODE 0: bf16 out, head layout [b,h,T,dh]
// MODE 1: bf16 out, transposed head layout [b,h,dh,T]   (for V)
// MODE 2: fp32 out, row-major [4096,2048]               (final projection)
template <int MODE>
__global__ __launch_bounds__(256) void gemm_bf16_k(const u16* __restrict__ A,
                                                   const u16* __restrict__ W,
                                                   void* __restrict__ out) {
  const int lane = threadIdx.x & 31;
  const int gw = (blockIdx.x * 256 + threadIdx.x) >> 5;
  const int nt = gw & 31;   // 2048 / 64 = 32 N-tiles
  const int mt = gw >> 5;   // 4096 / 32 = 128 M-tiles
  const int m0 = mt * 32, n0 = nt * 64;

  v8f acc[2][4];
#pragma unroll
  for (int i = 0; i < 2; ++i)
#pragma unroll
    for (int n = 0; n < 4; ++n) acc[i][n] = vzero8();

  for (int k = 0; k < 2048; k += 32) {
    v16bf a0 = load_fragA(A, 2048, m0, k, lane);
    v16bf a1 = load_fragA(A, 2048, m0 + 16, k, lane);
    v16bf bfr[4];
#pragma unroll
    for (int n = 0; n < 4; ++n) bfr[n] = load_fragB(W, 2048, n0 + n * 16, k, lane);
#pragma unroll
    for (int n = 0; n < 4; ++n) {
      acc[0][n] = wmma_bf16(a0, bfr[n], acc[0][n]);
      acc[1][n] = wmma_bf16(a1, bfr[n], acc[1][n]);
    }
  }

  // C/D layout: VGPR r, lanes 0-15 -> row r; lanes 16-31 -> row r+8
  const int rbase = (lane >> 4) * 8;
  const int cl = lane & 15;
#pragma unroll
  for (int i = 0; i < 2; ++i)
#pragma unroll
    for (int n = 0; n < 4; ++n)
#pragma unroll
      for (int r = 0; r < 8; ++r) {
        int m = m0 + i * 16 + rbase + r;
        int c = n0 + n * 16 + cl;
        float v = acc[i][n][r];
        if (MODE == 2) {
          ((float*)out)[(size_t)m * 2048 + c] = v;
        } else {
          int b = m >> 11, t = m & 2047;  // B=2, T=2048
          int h = c >> 7, d = c & 127;    // 16 heads, dh=128
          int bh = b * 16 + h;
          size_t idx = (MODE == 0) ? ((size_t)(bh * 2048 + t) * 128 + d)
                                   : ((size_t)(bh * 128 + d) * 2048 + t);
          ((u16*)out)[idx] = f32_to_bf16(v);
        }
      }
}

// ---------------------------------------------------------------- attention
// Block = 8 waves sharing one (b,h); wave w handles query tile qg*8+w.
// Per 32-key chunk: K tile (32x128, 8KB) and V tile (128x32 of Vt, 8KB) are
// staged into LDS once per block via async global->LDS copies (double
// buffered, ASYNCcnt + barrier handshake). K fragments are preloaded before
// the S chain; V fragments are preloaded before the softmax so their LDS
// latency hides under the exp/max VALU work. 16 v_wmma per chunk per wave.
__global__ __launch_bounds__(256) void attn_k(const u16* __restrict__ Q,
                                              const u16* __restrict__ K,
                                              const u16* __restrict__ Vt,
                                              u16* __restrict__ O) {
  __shared__ __align__(16) u16 kbuf[2][32 * 128];  // 2 x 8KB
  __shared__ __align__(16) u16 vbuf[2][128 * 32];  // 2 x 8KB
  __shared__ __align__(16) u16 pbuf[8][16 * 32];   // 1KB per wave
  const int lane = threadIdx.x & 31;
  const int wib = threadIdx.x >> 5;
  const int bh = blockIdx.x >> 4;  // 0..31
  const int qg = blockIdx.x & 15;  // query-tile group
  const int qt = qg * 8 + wib;     // this wave's 16-row query tile
  const u16* Qb = Q + ((size_t)bh * 2048 + qt * 16) * 128;
  const u16* Kb = K + (size_t)bh * 2048 * 128;
  const u16* Vb = Vt + (size_t)bh * 128 * 2048;
  u16* pb = &pbuf[wib][0];

  // stage one 32-key chunk of K and Vt into LDS (4 async b128 per wave)
  const unsigned long long kgb = (unsigned long long)(const char*)Kb;
  const unsigned long long vgb = (unsigned long long)(const char*)Vb;
  auto stage = [&](int kt0, int buf) {
    unsigned kl = (unsigned)(size_t)(void*)&kbuf[buf][0];
    unsigned long long kg = kgb + (size_t)kt0 * 256;  // contiguous 8KB
#pragma unroll
    for (int i = 0; i < 2; ++i) {
      unsigned off = wib * 1024 + i * 512 + lane * 16;
      async_ld_b128(kl + off, kg + off);
    }
    unsigned vl = (unsigned)(size_t)(void*)&vbuf[buf][0];
    unsigned long long vg = vgb + (size_t)kt0 * 2;  // 128 rows of 64B
#pragma unroll
    for (int i = 0; i < 2; ++i) {
      int row = wib * 16 + i * 8 + (lane >> 2);
      unsigned coff = (lane & 3) * 16;
      async_ld_b128(vl + row * 64 + coff, vg + (size_t)row * 4096 + coff);
    }
  };

  v16bf qf[4];
#pragma unroll
  for (int d = 0; d < 4; ++d) qf[d] = load_fragA(Qb, 128, 0, d * 32, lane);

  v8f acc[8];
#pragma unroll
  for (int n = 0; n < 8; ++n) acc[n] = vzero8();
  float mrow[8], psum[8];
#pragma unroll
  for (int r = 0; r < 8; ++r) {
    mrow[r] = -1e30f;
    psum[r] = 0.f;
  }

  const float sc = 0.08838834764831845f;  // 1/sqrt(128)
  const int rbase = (lane >> 4) * 8;
  const int cl = lane & 15;

  stage(0, 0);
  for (int c = 0; c < 64; ++c) {
    const int cur = c & 1;
    if (c < 63) {
      stage((c + 1) * 32, cur ^ 1);
      wait_async_le4();  // previous chunk's copies done (in-order)
    } else {
      wait_async_0();
    }
    __syncthreads();  // all waves' copies for chunk c are in LDS

    const u16* ktp = &kbuf[cur][0];
    const u16* vtp = &vbuf[cur][0];

    // preload all K fragments -> one ds clause, overlap with wmma chain
    v16bf kf[8];
#pragma unroll
    for (int d = 0; d < 4; ++d) {
      kf[d * 2 + 0] = load_fragB(ktp, 128, 0, d * 32, lane);
      kf[d * 2 + 1] = load_fragB(ktp, 128, 16, d * 32, lane);
    }
    v8f s0 = vzero8(), s1 = vzero8();
#pragma unroll
    for (int d = 0; d < 4; ++d) {
      s0 = wmma_bf16(qf[d], kf[d * 2 + 0], s0);
      s1 = wmma_bf16(qf[d], kf[d * 2 + 1], s1);
    }

    // preload V fragments now; their LDS latency hides under softmax VALU
    v16bf vf[8];
#pragma unroll
    for (int n = 0; n < 8; ++n) vf[n] = load_fragB(vtp, 32, n * 16, 0, lane);

#pragma unroll
    for (int r = 0; r < 8; ++r) {
      float a = s0[r] * sc, b = s1[r] * sc;
      float tm = fmaxf(a, b);  // rows live within a 16-lane group
      tm = fmaxf(tm, __shfl_xor(tm, 1, 32));
      tm = fmaxf(tm, __shfl_xor(tm, 2, 32));
      tm = fmaxf(tm, __shfl_xor(tm, 4, 32));
      tm = fmaxf(tm, __shfl_xor(tm, 8, 32));
      float nm = fmaxf(mrow[r], tm);
      float corr = __expf(mrow[r] - nm);
      mrow[r] = nm;
      psum[r] *= corr;
#pragma unroll
      for (int n = 0; n < 8; ++n) acc[n][r] *= corr;
      float p0 = __expf(a - nm);
      float p1 = __expf(b - nm);
      psum[r] += p0 + p1;  // per-lane partial row sum (corr row-uniform)
      pb[(rbase + r) * 32 + cl] = f32_to_bf16(p0);
      pb[(rbase + r) * 32 + 16 + cl] = f32_to_bf16(p1);
    }
    // D-layout -> A-layout transpose via per-wave LDS tile
    v16bf pf = load_fragA(pb, 32, 0, 0, lane);
#pragma unroll
    for (int n = 0; n < 8; ++n) acc[n] = wmma_bf16(pf, vf[n], acc[n]);
    __syncthreads();  // protect buf before next-next chunk overwrites it
  }

#pragma unroll
  for (int r = 0; r < 8; ++r) {
    float v = psum[r];
    v += __shfl_xor(v, 1, 32);
    v += __shfl_xor(v, 2, 32);
    v += __shfl_xor(v, 4, 32);
    v += __shfl_xor(v, 8, 32);
    psum[r] = 1.0f / v;
  }

  const int b = bh >> 4, h = bh & 15;
  const int t0 = qt * 16;
  u16* Ob = O + (size_t)b * 2048 * 2048;  // [B,T,D] bf16
#pragma unroll
  for (int n = 0; n < 8; ++n)
#pragma unroll
    for (int r = 0; r < 8; ++r) {
      int t = t0 + rbase + r;
      int c = h * 128 + n * 16 + cl;
      Ob[(size_t)t * 2048 + c] = f32_to_bf16(acc[n][r] * psum[r]);
    }
}

// ---------------------------------------------------------------- launch
extern "C" void kernel_launch(void* const* d_in, const int* in_sizes, int n_in,
                              void* d_out, int out_size, void* d_ws,
                              size_t ws_size, hipStream_t stream) {
  (void)in_sizes; (void)n_in; (void)out_size; (void)ws_size;
  const float* x = (const float*)d_in[0];
  const float* Wq = (const float*)d_in[1];
  const float* Wk = (const float*)d_in[2];
  const float* Wv = (const float*)d_in[3];
  const float* Wo = (const float*)d_in[4];
  float* out = (float*)d_out;

  const int NX = 4096 * 2048;  // 8388608
  const int NW = 2048 * 2048;  // 4194304

  u16* ws = (u16*)d_ws;
  u16* xb = ws;  // [4096,2048]
  u16* Wqb = xb + NX;
  u16* Wkb = Wqb + NW;
  u16* Wvb = Wkb + NW;
  u16* Wob = Wvb + NW;
  u16* Qh = Wob + NW;  // [b,h,T,dh]
  u16* Kh = Qh + NX;   // [b,h,T,dh]
  u16* Vth = Kh + NX;  // [b,h,dh,T]
  u16* Ab = Vth + NX;  // [B,T,D] attention output (bf16)

  cvt_bf16_k<<<(NX + 255) / 256, 256, 0, stream>>>(x, xb, NX);
  cvt_bf16_k<<<(NW + 255) / 256, 256, 0, stream>>>(Wq, Wqb, NW);
  cvt_bf16_k<<<(NW + 255) / 256, 256, 0, stream>>>(Wk, Wkb, NW);
  cvt_bf16_k<<<(NW + 255) / 256, 256, 0, stream>>>(Wv, Wvb, NW);
  cvt_bf16_k<<<(NW + 255) / 256, 256, 0, stream>>>(Wo, Wob, NW);

  // 4096 waves per GEMM -> 512 blocks of 8 waves
  gemm_bf16_k<0><<<512, 256, 0, stream>>>(xb, Wqb, Qh);
  gemm_bf16_k<0><<<512, 256, 0, stream>>>(xb, Wkb, Kh);
  gemm_bf16_k<1><<<512, 256, 0, stream>>>(xb, Wvb, Vth);

  // 32 (b,h) x 16 query-tile groups = 512 blocks of 8 waves
  attn_k<<<512, 256, 0, stream>>>(Qh, Kh, Vth, Ab);

  gemm_bf16_k<2><<<512, 256, 0, stream>>>(Ab, Wob, out);
}
